// AgesRRN_17995912970656
// MI455X (gfx1250) — matrix operand order repaired and checked by reading, not compiled, not run on hardware
//
#include <hip/hip_runtime.h>

// ---------------------------------------------------------------------------
// Recurrent Relational Network on MI455X (gfx1250), WMMA f16->f32 GEMMs.
// N_NODES=8, N_STEPS=8, bs=2048 -> N=16384 node states of 128 features.
// All GEMM configs are compile-time template specializations (no runtime
// nullable-pointer branching in the hot path).
// ---------------------------------------------------------------------------

#define NNODES 8
#define NSTEPS 8
#define BS     2048
#define NN     (BS * NNODES)   // 16384 nodes
#define F      128             // feature width

typedef __attribute__((ext_vector_type(16))) _Float16 v16h;
typedef __attribute__((ext_vector_type(8)))  _Float16 v8h;
typedef __attribute__((ext_vector_type(8)))  float    v8f;

// ----- A fragment: 16x32 f16 tile, row-major source with ld=128 ------------
// contiguous k order (this code's, not ISA table order): lane l (half hl=l>>4, r=l&15) holds row M=r,
//   elems 0..7  -> K = kbase + hl*16 + i
//   elems 8..15 -> K = kbase + hl*16 + i
__device__ __forceinline__ v16h load_a_frag(const _Float16* __restrict__ A,
                                            int rowbase, int kbase, int lane) {
    int r  = rowbase + (lane & 15);
    int hl = lane >> 4;
    const _Float16* p = A + (size_t)r * F + kbase + hl * 16;
    v8h lo = *(const v8h*)p;          // 16B load
    v8h hi = *(const v8h*)(p + 8);    // 16B load
    return __builtin_shufflevector(lo, hi,
        0,1,2,3,4,5,6,7,8,9,10,11,12,13,14,15);
}

// ----- B fragment: 32x16 f16 tile from LDS, pre-swizzled contiguous --------
__device__ __forceinline__ v16h load_b_frag(const _Float16* __restrict__ lds,
                                            int t, int c, int lane) {
    return *(const v16h*)(lds + (size_t)((((t << 2) + c) << 5) + lane) * 16);
}

// ----- stage a 128x128 f16 weight into B-fragment-swizzled LDS -------------
__device__ __forceinline__ void stage_w(const _Float16* __restrict__ W,
                                        _Float16* __restrict__ lds, int tid) {
    for (int t = tid; t < F * F; t += 256) {
        int k = t >> 7, n = t & 127;
        int tile = n >> 4, col = n & 15;
        int c = k >> 5,  kk = k & 31;
        int lane = ((kk >> 4) << 4) | col;
        lds[((((tile << 2) + c) << 5) + lane) * 16 + (kk & 15)] = W[t];
    }
}

// ---------------------------------------------------------------------------
// GEMM: out = act( A @ W  [+ A2 @ W2] [+ bias*biasScale] [+ Cin] )
// Fully specialized at compile time -> straight-line epilogue, no exec-mask
// branching around the WMMA chains.
// ---------------------------------------------------------------------------
template<bool HASA2, bool HASBIAS, bool HASCIN, bool RELU,
         bool O16, bool O16B, bool O32, bool NARROW>
__global__ __launch_bounds__(256)
void gemm128_t(const _Float16* __restrict__ A,  const _Float16* __restrict__ A2,
               const _Float16* __restrict__ W,  const _Float16* __restrict__ W2,
               const float* __restrict__ bias,  float biasScale,
               const float* __restrict__ Cin,
               _Float16* __restrict__ out16, _Float16* __restrict__ out16b,
               float* __restrict__ out32, int M, int ncolsStore)
{
    __shared__ _Float16 ldsW[(HASA2 ? 2 : 1) * F * F];

    const int tid = threadIdx.x;
    stage_w(W, &ldsW[0], tid);
    if constexpr (HASA2) stage_w(W2, &ldsW[F * F], tid);
    __syncthreads();

    const int wave = tid >> 5, lane = tid & 31;
    const int rowbase = blockIdx.x * 128 + wave * 16;
    if (rowbase >= M) return;

    v16h a[4];
#pragma unroll
    for (int c = 0; c < 4; ++c) a[c] = load_a_frag(A, rowbase, c * 32, lane);
    v16h a2[4];
    if constexpr (HASA2) {
#pragma unroll
        for (int c = 0; c < 4; ++c) a2[c] = load_a_frag(A2, rowbase, c * 32, lane);
    }

    const int colsel = lane & 15;
    const int rowoff = (lane >> 4) << 3;

#pragma unroll
    for (int t = 0; t < 8; ++t) {
        v8f acc = {};
#pragma unroll
        for (int c = 0; c < 4; ++c) {
            v16h b = load_b_frag(&ldsW[0], t, c, lane);
            acc = __builtin_amdgcn_wmma_f32_16x16x32_f16(
                false, a[c], false, b, (short)0, acc, false, false);
        }
        if constexpr (HASA2) {
#pragma unroll
            for (int c = 0; c < 4; ++c) {
                v16h b = load_b_frag(&ldsW[F * F], t, c, lane);
                acc = __builtin_amdgcn_wmma_f32_16x16x32_f16(
                    false, a2[c], false, b, (short)0, acc, false, false);
            }
        }
        const int ncol = (t << 4) | colsel;
        bool colok = true;
        if constexpr (NARROW) colok = (ncol < ncolsStore);
        float badd = 0.0f;
        if constexpr (HASBIAS) { if (colok) badd = bias[ncol] * biasScale; }
#pragma unroll
        for (int v = 0; v < 8; ++v) {
            int row = rowbase + rowoff + v;
            float val = acc[v] + badd;
            if constexpr (HASCIN) val += Cin[(size_t)row * F + ncol];
            if constexpr (RELU)   val  = fmaxf(val, 0.0f);
            if (colok) {
                if constexpr (O32)  out32 [(size_t)row * ncolsStore + ncol] = val;
                if constexpr (O16)  out16 [(size_t)row * F + ncol] = (_Float16)val;
                if constexpr (O16B) out16b[(size_t)row * F + ncol] = (_Float16)val;
            }
        }
    }
}

// ---------------------------------------------------------------------------
// Dual-output GEMM: O1 = A@W1 + bias1 ; O2 = A@W2  (shared A fragments).
// Used for the per-step P/Q message projections.
// ---------------------------------------------------------------------------
__global__ __launch_bounds__(256)
void gemm128_dual(const _Float16* __restrict__ A,
                  const _Float16* __restrict__ W1, const _Float16* __restrict__ W2,
                  const float* __restrict__ bias1,
                  float* __restrict__ O1, float* __restrict__ O2, int M)
{
    __shared__ _Float16 ldsW[2 * F * F];

    const int tid = threadIdx.x;
    stage_w(W1, &ldsW[0], tid);
    stage_w(W2, &ldsW[F * F], tid);
    __syncthreads();

    const int wave = tid >> 5, lane = tid & 31;
    const int rowbase = blockIdx.x * 128 + wave * 16;
    if (rowbase >= M) return;

    v16h a[4];
#pragma unroll
    for (int c = 0; c < 4; ++c) a[c] = load_a_frag(A, rowbase, c * 32, lane);

    const int colsel = lane & 15;
    const int rowoff = (lane >> 4) << 3;

#pragma unroll
    for (int t = 0; t < 8; ++t) {
        v8f acc1 = {}, acc2 = {};
#pragma unroll
        for (int c = 0; c < 4; ++c) {
            v16h b1 = load_b_frag(&ldsW[0], t, c, lane);
            acc1 = __builtin_amdgcn_wmma_f32_16x16x32_f16(
                false, a[c], false, b1, (short)0, acc1, false, false);
            v16h b2 = load_b_frag(&ldsW[F * F], t, c, lane);
            acc2 = __builtin_amdgcn_wmma_f32_16x16x32_f16(
                false, a[c], false, b2, (short)0, acc2, false, false);
        }
        const int ncol = (t << 4) | colsel;
        const float badd = bias1[ncol];
#pragma unroll
        for (int v = 0; v < 8; ++v) {
            size_t row = (size_t)(rowbase + rowoff + v);
            O1[row * F + ncol] = acc1[v] + badd;
            O2[row * F + ncol] = acc2[v];
        }
    }
}

// --- f32 -> f16 weight convert, pad columns to 128 with zeros --------------
__global__ void cvt_w_f16(const float* __restrict__ src, _Float16* __restrict__ dst,
                          int rows, int cols) {
    int idx = blockIdx.x * blockDim.x + threadIdx.x;
    if (idx >= rows * F) return;
    int r = idx >> 7, c = idx & 127;
    dst[idx] = (c < cols) ? (_Float16)src[(size_t)r * cols + c] : (_Float16)0.0f;
}

// --- Pre-MLP layer1: one-hot matmul == gather-sum of 5 rows of pre_w1 ------
__global__ void pre_feat_kernel(const int* __restrict__ src, const int* __restrict__ tgt,
                                const int* __restrict__ typ, const int* __restrict__ dif,
                                const int* __restrict__ qst,
                                const float* __restrict__ w1, const float* __restrict__ b1,
                                _Float16* __restrict__ hid1) {
    int idx = blockIdx.x * blockDim.x + threadIdx.x;
    if (idx >= NN * F) return;
    int f = idx & 127;
    int n = idx >> 7;
    int g = n >> 3;
    float v = b1[f]
            + w1[(size_t)(      src[n]) * F + f]
            + w1[(size_t)(  8 + tgt[n]) * F + f]
            + w1[(size_t)( 16 + typ[n]) * F + f]
            + w1[(size_t)( 19 + dif[n]) * F + f]
            + w1[(size_t)(119 + qst[g]) * F + f];
    hid1[idx] = (_Float16)fmaxf(v, 0.0f);
}

// --- Edge combine: s1[i] = sum_{j!=i in graph} relu(P[i] + Q[j]) -----------
// (msg_b1 already folded into P.)
__global__ void edge_combine_kernel(const float* __restrict__ P, const float* __restrict__ Q,
                                    _Float16* __restrict__ s1) {
    int idx = blockIdx.x * blockDim.x + threadIdx.x;
    if (idx >= NN * F) return;
    int f = idx & 127;
    int n = idx >> 7;
    int g = n >> 3, i = n & 7;
    float p = P[idx];
    const float* qb = Q + ((size_t)(g << 3) << 7) + f;
    float acc = 0.0f;
#pragma unroll
    for (int j = 0; j < 8; ++j) {
        if (j == i) continue;
        float v = p + qb[(size_t)j * F];
        acc += (v > 0.0f) ? v : 0.0f;
    }
    s1[idx] = (_Float16)acc;
}

// --- Graph pooling: pooled[g] = sum over the graph's 8 nodes of h_new ------
__global__ void pool_kernel(const float* __restrict__ hnew, _Float16* __restrict__ pooled) {
    int idx = blockIdx.x * blockDim.x + threadIdx.x;
    if (idx >= BS * F) return;
    int f = idx & 127;
    int g = idx >> 7;
    const float* hb = hnew + ((size_t)(g << 3) << 7) + f;
    float s = 0.0f;
#pragma unroll
    for (int i = 0; i < 8; ++i) s += hb[(size_t)i * F];
    pooled[idx] = (_Float16)s;
}

// ---------------------------------------------------------------------------
extern "C" void kernel_launch(void* const* d_in, const int* in_sizes, int n_in,
                              void* d_out, int out_size, void* d_ws, size_t ws_size,
                              hipStream_t stream) {
    (void)in_sizes; (void)n_in; (void)out_size; (void)ws_size;

    const int*   sources = (const int*)  d_in[0];
    const int*   targets = (const int*)  d_in[1];
    const int*   types   = (const int*)  d_in[2];
    const int*   diffs   = (const int*)  d_in[3];
    const int*   quest   = (const int*)  d_in[4];
    const float* pre_w1  = (const float*)d_in[5];
    const float* pre_b1  = (const float*)d_in[6];
    const float* pre_w2  = (const float*)d_in[7];
    const float* pre_b2  = (const float*)d_in[8];
    const float* msg_w1  = (const float*)d_in[9];
    const float* msg_b1  = (const float*)d_in[10];
    const float* msg_w2  = (const float*)d_in[11];
    const float* msg_b2  = (const float*)d_in[12];
    const float* node_w1 = (const float*)d_in[13];
    const float* node_b1 = (const float*)d_in[14];
    const float* node_w2 = (const float*)d_in[15];
    const float* node_b2 = (const float*)d_in[16];
    const float* out_w1  = (const float*)d_in[17];
    const float* out_b1  = (const float*)d_in[18];
    const float* out_w2  = (const float*)d_in[19];
    const float* out_b2  = (const float*)d_in[20];
    float* logits = (float*)d_out;

    // ---- workspace carve-up (256B aligned) --------------------------------
    char* ws = (char*)d_ws;
    size_t off = 0;
    auto take = [&](size_t bytes) -> void* {
        void* p = ws + off;
        off += (bytes + 255) & ~(size_t)255;
        return p;
    };
    const size_t WB = (size_t)F * F * sizeof(_Float16);   // 32 KB per weight
    _Float16* wpre2 = (_Float16*)take(WB);
    _Float16* wm1a  = (_Float16*)take(WB);
    _Float16* wm1b  = (_Float16*)take(WB);
    _Float16* wm2   = (_Float16*)take(WB);
    _Float16* wn1a  = (_Float16*)take(WB);
    _Float16* wn1b  = (_Float16*)take(WB);
    _Float16* wn1c  = (_Float16*)take(WB);
    _Float16* wn2   = (_Float16*)take(WB);
    _Float16* wo1   = (_Float16*)take(WB);
    _Float16* wo2   = (_Float16*)take(WB);  // 128x100 padded to 128x128

    const size_t NF16 = (size_t)NN * F * sizeof(_Float16);
    const size_t NF32 = (size_t)NN * F * sizeof(float);
    _Float16* hid1   = (_Float16*)take(NF16);
    _Float16* x16    = (_Float16*)take(NF16);
    _Float16* hA     = (_Float16*)take(NF16);
    _Float16* hB     = (_Float16*)take(NF16);
    _Float16* s1     = (_Float16*)take(NF16);
    _Float16* magg   = (_Float16*)take(NF16);
    _Float16* n1     = (_Float16*)take(NF16);
    float*    xb     = (float*)take(NF32);
    float*    Pm     = (float*)take(NF32);
    float*    Qm     = (float*)take(NF32);
    float*    hnew32 = (float*)take(NF32);
    _Float16* pooled = (_Float16*)take((size_t)BS * F * sizeof(_Float16));
    _Float16* ohid   = (_Float16*)take((size_t)BS * F * sizeof(_Float16));

    const dim3 B256(256);
    const dim3 Gcvt(64);                 // 128*128/256
    const dim3 Gelem(NN * F / 256);      // 8192
    const dim3 Gpool(BS * F / 256);      // 1024
    const dim3 GgemmN(NN / 128);         // 128
    const dim3 GgemmB(BS / 128);         // 16

    // ---- weight conversion f32 -> f16 (B layout = [k][n] row-major) -------
    cvt_w_f16<<<Gcvt, B256, 0, stream>>>(pre_w2,              wpre2, 128, 128);
    cvt_w_f16<<<Gcvt, B256, 0, stream>>>(msg_w1,              wm1a,  128, 128);
    cvt_w_f16<<<Gcvt, B256, 0, stream>>>(msg_w1 + 128 * 128,  wm1b,  128, 128);
    cvt_w_f16<<<Gcvt, B256, 0, stream>>>(msg_w2,              wm2,   128, 128);
    cvt_w_f16<<<Gcvt, B256, 0, stream>>>(node_w1,             wn1a,  128, 128);
    cvt_w_f16<<<Gcvt, B256, 0, stream>>>(node_w1 + 128 * 128, wn1b,  128, 128);
    cvt_w_f16<<<Gcvt, B256, 0, stream>>>(node_w1 + 256 * 128, wn1c,  128, 128);
    cvt_w_f16<<<Gcvt, B256, 0, stream>>>(node_w2,             wn2,   128, 128);
    cvt_w_f16<<<Gcvt, B256, 0, stream>>>(out_w1,              wo1,   128, 128);
    cvt_w_f16<<<Gcvt, B256, 0, stream>>>(out_w2,              wo2,   128, 100);

    // ---- pre-MLP: hid1 = relu(onehot @ pre_w1 + b1)  (row-gather sum) -----
    pre_feat_kernel<<<Gelem, B256, 0, stream>>>(sources, targets, types, diffs,
                                                quest, pre_w1, pre_b1, hid1);

    // x = hid1 @ pre_w2 + b2 ; h0 = x (dup store into hA)
    gemm128_t<false, true, false, false, true, true, false, false>
        <<<GgemmN, B256, 0, stream>>>(hid1, nullptr, wpre2, nullptr,
            pre_b2, 1.0f, nullptr, x16, hA, nullptr, NN, F);
    // xb = x @ node_w1[0:128] + node_b1   (step-invariant)
    gemm128_t<false, true, false, false, false, false, true, false>
        <<<GgemmN, B256, 0, stream>>>(x16, nullptr, wn1a, nullptr,
            node_b1, 1.0f, nullptr, nullptr, nullptr, xb, NN, F);

    _Float16* hcur = hA;
    _Float16* hnxt = hB;
    for (int step = 0; step < NSTEPS; ++step) {
        // P = h @ msg_w1[0:128] + msg_b1 ;  Q = h @ msg_w1[128:256]  (fused)
        gemm128_dual<<<GgemmN, B256, 0, stream>>>(hcur, wm1a, wm1b,
            msg_b1, Pm, Qm, NN);
        // s1[i] = sum_{j!=i} relu(P[i] + Q[j])
        edge_combine_kernel<<<Gelem, B256, 0, stream>>>(Pm, Qm, s1);
        // m_agg = s1 @ msg_w2 + 7*msg_b2   (layer-2 commutes with segment_sum)
        gemm128_t<false, true, false, false, true, false, false, false>
            <<<GgemmN, B256, 0, stream>>>(s1, nullptr, wm2, nullptr,
                msg_b2, 7.0f, nullptr, magg, nullptr, nullptr, NN, F);
        // n1 = relu(xb + h @ Wb + m_agg @ Wc)
        gemm128_t<true, false, true, true, true, false, false, false>
            <<<GgemmN, B256, 0, stream>>>(hcur, magg, wn1b, wn1c,
                nullptr, 1.0f, xb, n1, nullptr, nullptr, NN, F);
        // h_new = n1 @ node_w2 + node_b2  -> f16 (next h) and f32 (pooling)
        gemm128_t<false, true, false, false, true, false, true, false>
            <<<GgemmN, B256, 0, stream>>>(n1, nullptr, wn2, nullptr,
                node_b2, 1.0f, nullptr, hnxt, nullptr, hnew32, NN, F);
        // pooled[g] = sum over 8 nodes of h_new
        pool_kernel<<<Gpool, B256, 0, stream>>>(hnew32, pooled);
        // out MLP: ohid = relu(pooled @ out_w1 + b1)
        gemm128_t<false, true, false, true, true, false, false, false>
            <<<GgemmB, B256, 0, stream>>>(pooled, nullptr, wo1, nullptr,
                out_b1, 1.0f, nullptr, ohid, nullptr, nullptr, BS, F);
        // logits = ohid @ out_w2 + b2  (narrow 100-col store)
        gemm128_t<false, true, false, false, false, false, true, true>
            <<<GgemmB, B256, 0, stream>>>(ohid, nullptr, wo2, nullptr,
                out_b2, 1.0f, nullptr, nullptr, nullptr,
                logits + (size_t)step * BS * 100, BS, 100);

        _Float16* t = hcur; hcur = hnxt; hnxt = t;
    }
}
